// ConvAttention_84344567759681
// MI455X (gfx1250) — matrix operand order repaired and checked by reading, not compiled, not run on hardware
//
#include <hip/hip_runtime.h>
#include <hip/hip_bf16.h>

// Problem constants (match reference setup_inputs)
#define BATCH 8
#define EMB   256
#define SEQ   4096
#define HEADS 8
#define GC    (EMB/HEADS)   // 32 channels per group
#define KW    3

typedef _Float16 v16h __attribute__((ext_vector_type(16)));
typedef float    v8f  __attribute__((ext_vector_type(8)));

// ---------------------------------------------------------------------------
// CDNA5 async global->LDS copy helpers (ASYNCcnt-tracked DMA path).
// 16 bytes per lane per issue; LDS dest address is a per-lane VGPR offset.
// ---------------------------------------------------------------------------
__device__ __forceinline__ void async_copy_b128(unsigned lds_off, const void* gaddr) {
    asm volatile("global_load_async_to_lds_b128 %0, %1, off"
                 :: "v"(lds_off), "v"(gaddr) : "memory");
}
__device__ __forceinline__ void wait_async_le8() {
    asm volatile("s_wait_asynccnt 0x8" ::: "memory");
}
__device__ __forceinline__ void wait_async_0() {
    asm volatile("s_wait_asynccnt 0x0" ::: "memory");
}
__device__ __forceinline__ unsigned lds_off_of(const void* p) {
    // generic pointers to LDS carry the byte offset in the low 32 bits
    return (unsigned)(size_t)p;
}

// ---------------------------------------------------------------------------
// Kernel 1: grouped conv1d (q,k,v projections) fp32 -> f16
//   q,k written [B,T,E] (row-major over E: A/B-operand friendly)
//   v  written [B,E,T] (transposed: B-operand friendly for P·V)
// ---------------------------------------------------------------------------
#define CTB 128  // t-tile per block

__global__ __launch_bounds__(256) void conv_qkv_kernel(
    const float* __restrict__ x,
    const float* __restrict__ wq, const float* __restrict__ bq,
    const float* __restrict__ wk, const float* __restrict__ bk,
    const float* __restrict__ wv, const float* __restrict__ bv,
    _Float16* __restrict__ qo, _Float16* __restrict__ ko,
    _Float16* __restrict__ vo)
{
    __shared__ float xs[GC][CTB + 2];

    const int b  = blockIdx.z;
    const int g  = blockIdx.y;
    const int t0 = blockIdx.x * CTB;
    const int tid = threadIdx.x;

    const float* xb = x + ((size_t)b * EMB + (size_t)g * GC) * SEQ;
    for (int i = tid; i < GC * (CTB + 2); i += 256) {
        int c = i / (CTB + 2), j = i % (CTB + 2);
        int t = t0 - 1 + j;
        xs[c][j] = (t >= 0 && t < SEQ) ? xb[(size_t)c * SEQ + t] : 0.0f;
    }
    __syncthreads();

    const float* W[3]  = {wq, wk, wv};
    const float* Bi[3] = {bq, bk, bv};

    const int o  = tid & 31;   // output channel within group
    const int tb = tid >> 5;   // 0..7 t-phase

    for (int p = 0; p < 3; ++p) {
        const float* wp = W[p] + ((size_t)(g * GC + o)) * (GC * KW);
        float wr[GC * KW];
        #pragma unroll
        for (int i = 0; i < GC * KW; ++i) wr[i] = wp[i];
        const float bias = Bi[p][g * GC + o];
        const int oc = g * GC + o;

        #pragma unroll 1
        for (int it = 0; it < CTB / 8; ++it) {
            const int tl = tb + it * 8;
            float acc = bias;
            #pragma unroll
            for (int c = 0; c < GC; ++c) {
                acc += xs[c][tl + 0] * wr[c * 3 + 0];
                acc += xs[c][tl + 1] * wr[c * 3 + 1];
                acc += xs[c][tl + 2] * wr[c * 3 + 2];
            }
            const int t = t0 + tl;
            const _Float16 hv = (_Float16)acc;
            if (p == 0)      qo[((size_t)b * SEQ + t) * EMB + oc] = hv;
            else if (p == 1) ko[((size_t)b * SEQ + t) * EMB + oc] = hv;
            else             vo[((size_t)b * EMB + oc) * SEQ + t] = hv;
        }
    }
}

// ---------------------------------------------------------------------------
// Kernel 2: w_fc fp32 -> f16 (one shot, 64K elements)
// ---------------------------------------------------------------------------
__global__ __launch_bounds__(256) void cvt_wfc_kernel(
    const float* __restrict__ w, _Float16* __restrict__ wh)
{
    int i = blockIdx.x * 256 + threadIdx.x;
    wh[i] = (_Float16)w[i];
}

// ---------------------------------------------------------------------------
// Kernel 3: flash attention.
//   Block = 8 waves = 8 consecutive 16-query tiles of one batch; all waves
//   share the K/V stream, which is staged cooperatively into LDS with
//   double-buffered global_load_async_to_lds_b128 (ASYNCcnt), one chunk ahead.
//   S = (Q K^T)/16 via v_wmma_f32_16x16x32_f16; online softmax in C-layout
//   registers; P transposed C->A through a per-wave LDS buffer; O += P V.
// ---------------------------------------------------------------------------
__global__ __launch_bounds__(256) void flash_attn_kernel(
    const _Float16* __restrict__ Q, const _Float16* __restrict__ Kt,
    const _Float16* __restrict__ Vt, _Float16* __restrict__ Og)
{
    __shared__ _Float16 kbuf[2][32][EMB];     // 2 x 16 KB: K rows  [s-local][e]
    __shared__ _Float16 vbuf[2][EMB][32];     // 2 x 16 KB: V cols  [e][s-local]
    __shared__ _Float16 plds[8][16][34];      // per-wave P transpose buffer

    const int tid  = threadIdx.x;
    const int wave = tid >> 5;
    const int lane = tid & 31;
    const int m16  = lane & 15;
    const int h    = lane >> 4;

    const int gw = blockIdx.x * 8 + wave;        // global q-tile id
    const int b  = gw >> 8;                      // 256 tiles per batch
    const int q0 = (gw & 255) << 4;

    const _Float16* Qb = Q  + (size_t)b * SEQ * EMB;
    const _Float16* Kb = Kt + (size_t)b * SEQ * EMB;
    const _Float16* Vb = Vt + (size_t)b * EMB * SEQ;

    // ---- Q A-fragments (8 chunks of K=32 over E=256) ----
    v16h qa[8];
    #pragma unroll
    for (int kc = 0; kc < 8; ++kc) {
        const _Float16* r = Qb + (size_t)(q0 + m16) * EMB + kc * 32;
        #pragma unroll
        for (int j = 0; j < 8; ++j) {
            qa[kc][j]     = r[16 * h + j];
            qa[kc][j + 8] = r[16 * h + 8 + j];
        }
    }

    float mrow[8], lrow[8];
    v8f   oc[16];
    #pragma unroll
    for (int r = 0; r < 8; ++r) { mrow[r] = -1e30f; lrow[r] = 0.0f; }
    #pragma unroll
    for (int ec = 0; ec < 16; ++ec)
        #pragma unroll
        for (int r = 0; r < 8; ++r) oc[ec][r] = 0.0f;

    const float scale = 0.0625f;   // 1/sqrt(256)
    const int   NCH   = SEQ / 32;  // 128 key chunks

    // ---- async stage of key chunk `c` into buffer `bi` (8 b128 per thread) --
    auto stage = [&](int s0, int bi) {
        // K: rows s0..s0+31 are 16 KB contiguous in [T,E]
        const char* gk = (const char*)(Kb + (size_t)s0 * EMB);
        unsigned    lk = lds_off_of(&kbuf[bi][0][0]) + tid * 16;
        #pragma unroll
        for (int stp = 0; stp < 4; ++stp)
            async_copy_b128(lk + stp * 4096, gk + tid * 16 + stp * 4096);
        // V: row e = tid contributes 64 contiguous bytes Vt[e][s0..s0+31]
        const char* gv = (const char*)(Vb + (size_t)tid * SEQ + s0);
        unsigned    lv = lds_off_of(&vbuf[bi][0][0]) + tid * 64;
        #pragma unroll
        for (int stp = 0; stp < 4; ++stp)
            async_copy_b128(lv + stp * 16, gv + stp * 16);
    };

    stage(0, 0);   // prologue: chunk 0 in flight

    #pragma unroll 1
    for (int it = 0; it < NCH; ++it) {
        const int s0 = it * 32;
        const int bi = it & 1;

        if (it + 1 < NCH) { stage(s0 + 32, bi ^ 1); wait_async_le8(); }
        else              { wait_async_0(); }
        __syncthreads();   // all waves' chunk-`it` copies visible

        // ---- scores: two adjacent 16-wide key tiles from LDS ----
        v8f s_0, s_1;
        #pragma unroll
        for (int r = 0; r < 8; ++r) { s_0[r] = 0.0f; s_1[r] = 0.0f; }
        #pragma unroll
        for (int kc = 0; kc < 8; ++kc) {
            v16h kb0, kb1;
            const _Float16* r0 = &kbuf[bi][m16][kc * 32 + 16 * h];
            const _Float16* r1 = &kbuf[bi][16 + m16][kc * 32 + 16 * h];
            #pragma unroll
            for (int j = 0; j < 16; ++j) { kb0[j] = r0[j]; kb1[j] = r1[j]; }
            s_0 = __builtin_amdgcn_wmma_f32_16x16x32_f16(false, qa[kc], false, kb0,
                                                         (short)0, s_0, false, false);
            s_1 = __builtin_amdgcn_wmma_f32_16x16x32_f16(false, qa[kc], false, kb1,
                                                         (short)0, s_1, false, false);
        }

        // ---- online softmax update ----
        float rm[8];
        #pragma unroll
        for (int r = 0; r < 8; ++r) {
            s_0[r] *= scale; s_1[r] *= scale;
            rm[r] = fmaxf(s_0[r], s_1[r]);
        }
        #pragma unroll
        for (int msk = 1; msk < 16; msk <<= 1)
            #pragma unroll
            for (int r = 0; r < 8; ++r)
                rm[r] = fmaxf(rm[r], __shfl_xor(rm[r], msk, 32));

        float corr[8], rs[8];
        #pragma unroll
        for (int r = 0; r < 8; ++r) {
            float mn = fmaxf(mrow[r], rm[r]);
            corr[r]  = __expf(mrow[r] - mn);
            mrow[r]  = mn;
            s_0[r] = __expf(s_0[r] - mn);
            s_1[r] = __expf(s_1[r] - mn);
            rs[r]  = s_0[r] + s_1[r];
        }
        #pragma unroll
        for (int msk = 1; msk < 16; msk <<= 1)
            #pragma unroll
            for (int r = 0; r < 8; ++r)
                rs[r] += __shfl_xor(rs[r], msk, 32);
        #pragma unroll
        for (int r = 0; r < 8; ++r)
            lrow[r] = lrow[r] * corr[r] + rs[r];

        #pragma unroll
        for (int ec = 0; ec < 16; ++ec)
            #pragma unroll
            for (int r = 0; r < 8; ++r) oc[ec][r] *= corr[r];

        // ---- transpose P: C-layout -> A-layout via LDS (f16) ----
        __syncthreads();
        #pragma unroll
        for (int r = 0; r < 8; ++r) {
            plds[wave][r + 8 * h][m16]      = (_Float16)s_0[r];
            plds[wave][r + 8 * h][16 + m16] = (_Float16)s_1[r];
        }
        __syncthreads();
        v16h pa;
        #pragma unroll
        for (int j = 0; j < 8; ++j) {
            pa[j]     = plds[wave][m16][16 * h + j];
            pa[j + 8] = plds[wave][m16][16 * h + 8 + j];
        }

        // ---- O += P * V (16 e-chunks) from LDS ----
        #pragma unroll
        for (int ec = 0; ec < 16; ++ec) {
            v16h vb;
            const _Float16* vr = &vbuf[bi][ec * 16 + m16][16 * h];
            #pragma unroll
            for (int j = 0; j < 16; ++j) vb[j] = vr[j];
            oc[ec] = __builtin_amdgcn_wmma_f32_16x16x32_f16(false, pa, false, vb,
                                                            (short)0, oc[ec], false, false);
        }

        __syncthreads();   // all reads of buffer `bi` done before it is refilled
    }

    // ---- finalize (1/l) and store O as f16 [B,T,E] ----
    #pragma unroll
    for (int r = 0; r < 8; ++r) lrow[r] = 1.0f / lrow[r];
    #pragma unroll
    for (int ec = 0; ec < 16; ++ec)
        #pragma unroll
        for (int r = 0; r < 8; ++r) {
            float v = oc[ec][r] * lrow[r];
            Og[((size_t)b * SEQ + q0 + r + 8 * h) * EMB + ec * 16 + m16] = (_Float16)v;
        }
}

// ---------------------------------------------------------------------------
// Kernel 4: FC  out[b,t,o] = sum_e O[b,t,e] * Wfc[o,e] + bfc[o], f32 output.
// ---------------------------------------------------------------------------
__global__ __launch_bounds__(256) void fc_out_kernel(
    const _Float16* __restrict__ O, const _Float16* __restrict__ Wfc16,
    const float* __restrict__ bfc, float* __restrict__ out)
{
    const int tid  = threadIdx.x;
    const int wave = tid >> 5;
    const int lane = tid & 31;
    const int m16  = lane & 15;
    const int h    = lane >> 4;

    const int rt = blockIdx.x;          // 0..2047 row tiles
    const int b  = rt >> 8;
    const int q0 = (rt & 255) << 4;

    const _Float16* Ob = O + ((size_t)b * SEQ + q0) * EMB;

    v16h a[8];
    #pragma unroll
    for (int kc = 0; kc < 8; ++kc) {
        const _Float16* r = Ob + (size_t)m16 * EMB + kc * 32;
        #pragma unroll
        for (int j = 0; j < 8; ++j) {
            a[kc][j]     = r[16 * h + j];
            a[kc][j + 8] = r[16 * h + 8 + j];
        }
    }

    #pragma unroll
    for (int t = 0; t < 2; ++t) {
        const int o0 = (wave * 2 + t) * 16;
        v8f acc;
        #pragma unroll
        for (int r = 0; r < 8; ++r) acc[r] = 0.0f;
        #pragma unroll
        for (int kc = 0; kc < 8; ++kc) {
            v16h wb;   // B[k=e][n=o]: lane n=o, contiguous e run of w_fc row
            const _Float16* wr = Wfc16 + (size_t)(o0 + m16) * EMB + kc * 32 + 16 * h;
            #pragma unroll
            for (int j = 0; j < 16; ++j) wb[j] = wr[j];
            acc = __builtin_amdgcn_wmma_f32_16x16x32_f16(false, a[kc], false, wb,
                                                         (short)0, acc, false, false);
        }
        const float bias = bfc[o0 + m16];
        #pragma unroll
        for (int r = 0; r < 8; ++r)
            out[((size_t)b * SEQ + q0 + r + 8 * h) * EMB + o0 + m16] = acc[r] + bias;
    }
}

// ---------------------------------------------------------------------------
extern "C" void kernel_launch(void* const* d_in, const int* in_sizes, int n_in,
                              void* d_out, int out_size, void* d_ws, size_t ws_size,
                              hipStream_t stream)
{
    const float* x    = (const float*)d_in[0];
    const float* wq   = (const float*)d_in[1];
    const float* bq   = (const float*)d_in[2];
    const float* wk   = (const float*)d_in[3];
    const float* bk   = (const float*)d_in[4];
    const float* wv   = (const float*)d_in[5];
    const float* bv   = (const float*)d_in[6];
    const float* wfc  = (const float*)d_in[7];
    const float* bfc  = (const float*)d_in[8];
    float* out = (float*)d_out;

    const size_t NBTE = (size_t)BATCH * SEQ * EMB;   // 8.4M elements
    _Float16* qws   = (_Float16*)d_ws;               // [B,T,E] f16
    _Float16* kws   = qws + NBTE;                    // [B,T,E] f16
    _Float16* vws   = kws + NBTE;                    // [B,E,T] f16 (transposed)
    _Float16* ows   = vws + NBTE;                    // [B,T,E] f16 attention out
    _Float16* wfc16 = ows + NBTE;                    // [E,E]   f16 weights
    // total scratch: 64 MB + 128 KB

    dim3 cgrid(SEQ / CTB, HEADS, BATCH);             // 32 x 8 x 8
    conv_qkv_kernel<<<cgrid, 256, 0, stream>>>(x, wq, bq, wk, bk, wv, bv,
                                               qws, kws, vws);

    cvt_wfc_kernel<<<dim3(EMB * EMB / 256), 256, 0, stream>>>(wfc, wfc16);

    flash_attn_kernel<<<dim3((BATCH * SEQ / 16) / 8), 256, 0, stream>>>(
        qws, kws, vws, ows);

    fc_out_kernel<<<dim3(BATCH * SEQ / 16), 256, 0, stream>>>(ows, wfc16, bfc, out);
}
